// VisionTransformer_52458730553958
// MI455X (gfx1250) — matrix operand order repaired and checked by reading, not compiled, not run on hardware
//
#include <hip/hip_runtime.h>

// ---------------------------------------------------------------------------
// Types for CDNA5 WMMA
// ---------------------------------------------------------------------------
typedef __attribute__((ext_vector_type(16))) __bf16 v16bf;
typedef __attribute__((ext_vector_type(8)))  float  v8f;

union BFrag { v16bf v; unsigned u[8]; };

__device__ inline v8f wmma_bf16(const BFrag& a, const BFrag& b, v8f c) {
  return __builtin_amdgcn_wmma_f32_16x16x32_bf16(false, a.v, false, b.v,
                                                 (short)0, c, false, false);
}

__device__ inline unsigned short f2bf(float f) {
  union { float f; unsigned u; } v; v.f = f;
  unsigned r = v.u + 0x7FFFu + ((v.u >> 16) & 1u);   // round-to-nearest-even
  return (unsigned short)(r >> 16);
}
__device__ inline float bf2f(unsigned short h) {
  union { unsigned u; float f; } v; v.u = ((unsigned)h) << 16;
  return v.f;
}
__device__ inline unsigned pk2(float a, float b) {
  return (unsigned)f2bf(a) | ((unsigned)f2bf(b) << 16);
}

// DPP16 lane swaps (VALU, no LDS traffic): xor1/2/4/8 within 16-lane rows
#define DPP_XOR1 0xB1    // quad_perm(1,0,3,2)
#define DPP_XOR2 0x4E    // quad_perm(2,3,0,1)
#define DPP_XOR4 0x141   // row_half_mirror
#define DPP_XOR8 0x140   // row_mirror
template<int CTRL>
__device__ inline float dppf(float x) {
  union { float f; int i; } a, b;
  a.f = x;
  b.i = __builtin_amdgcn_mov_dpp(a.i, CTRL, 0xF, 0xF, true);
  return b.f;
}
__device__ inline float half16_max(float x) {   // max across lanes^{1,2,4,8}
  x = fmaxf(x, dppf<DPP_XOR1>(x));
  x = fmaxf(x, dppf<DPP_XOR2>(x));
  x = fmaxf(x, dppf<DPP_XOR4>(x));
  x = fmaxf(x, dppf<DPP_XOR8>(x));
  return x;
}
__device__ inline float half16_sum(float x) {
  x += dppf<DPP_XOR1>(x);
  x += dppf<DPP_XOR2>(x);
  x += dppf<DPP_XOR4>(x);
  x += dppf<DPP_XOR8>(x);
  return x;
}

#define NTOK 2304
#define DMODEL 1024
#define DFF 4096

// ---------------------------------------------------------------------------
// im2col: 16x16 patches of the 4 images -> (2304 x 768) bf16
// k = c*256 + py*16 + px  (matches patch_w layout (D,3,16,16))
// ---------------------------------------------------------------------------
__global__ void im2col_k(const float* __restrict__ i0, const float* __restrict__ i1,
                         const float* __restrict__ i2, const float* __restrict__ i3,
                         unsigned short* __restrict__ imb) {
  int idx = blockIdx.x * blockDim.x + threadIdx.x;   // over 2304*768
  if (idx >= NTOK * 768) return;
  int k = idx % 768, t = idx / 768;
  int c = k >> 8, rem = k & 255, py = rem >> 4, px = rem & 15;
  const float* img; int W, H, loc;
  if (t < 512)       { img = i0; W = 512; H = 256; loc = t; }
  else if (t < 1536) { img = i1; W = 512; H = 512; loc = t - 512; }
  else if (t < 2048) { img = i2; W = 256; H = 512; loc = t - 1536; }
  else               { img = i3; W = 256; H = 256; loc = t - 2048; }
  int wp = W >> 4;
  int y = loc / wp, x = loc % wp;
  float v = img[(size_t)c * H * W + (size_t)(y * 16 + py) * W + (x * 16 + px)];
  imb[idx] = f2bf(v);
}

// ---------------------------------------------------------------------------
// Generic GEMM:  C[M x Nc] = A[M x K](bf16) * W[Nc x K](f32->bf16)^T  (+resid)
// 128 threads (4 waves, 2x2), block tile 64x64, K-chunk 64 (8 wmma/wave per
// barrier round). LDS row stride = 36 dwords (32 data + 4 pad): 36r mod 64
// distinct for r=0..15 -> conflict-free b128 fragment loads, 16B aligned.
// ---------------------------------------------------------------------------
#define LSTR 36

template<int OUT_BF16, int HAS_RESID>
__global__ __launch_bounds__(128)
void gemm_bf16_wmma(const unsigned short* __restrict__ A,
                    const float* __restrict__ W,
                    const float* __restrict__ resid,
                    void* __restrict__ Cp,
                    int M, int Nc, int K) {
  __shared__ unsigned As[64 * LSTR];
  __shared__ unsigned Ws[64 * LSTR];
  int tid = threadIdx.x, lane = tid & 31, wid = tid >> 5;
  int wm = (wid >> 1) * 32, wn = (wid & 1) * 32;
  int m0 = blockIdx.y * 64, n0 = blockIdx.x * 64;
  int l15 = lane & 15;
  int half4 = (lane >= 16) ? 4 : 0;   // A koff (dwords)
  int half8 = (lane >= 16) ? 8 : 0;   // unused (B also uses half4)

  v8f acc[2][2] = {};
  const unsigned* A32 = (const unsigned*)A;
  int K2 = K >> 1;

  for (int k0 = 0; k0 < K; k0 += 64) {
    __syncthreads();
    // A tile: 64 rows x 64 bf16 = 512 uint4, 4 per thread
    #pragma unroll
    for (int it = 0; it < 4; ++it) {
      int idx = tid + it * 128;
      int r = idx >> 3, s = idx & 7;
      const uint4* src = (const uint4*)(A32 + (size_t)(m0 + r) * K2 + (k0 >> 1)) + s;
      uint4 v = *src;
      *(uint4*)(&As[r * LSTR + s * 4]) = v;
      if (k0 + 64 < K) __builtin_prefetch((const char*)(src + 16), 0, 0); // two k-slices ahead
    }
    // W tile: 64 rows x 64 f32 -> bf16 (1024 float4, 8 per thread)
    #pragma unroll
    for (int it = 0; it < 8; ++it) {
      int idx = tid + it * 128;
      int r = idx >> 4, s = idx & 15;
      const float4* src = (const float4*)(W + (size_t)(n0 + r) * K + k0) + s;
      float4 v = *src;
      Ws[r * LSTR + s * 2]     = pk2(v.x, v.y);
      Ws[r * LSTR + s * 2 + 1] = pk2(v.z, v.w);
      if (k0 + 64 < K) __builtin_prefetch((const char*)(src + 16), 0, 0);
    }
    __syncthreads();

    #pragma unroll
    for (int c = 0; c < 2; ++c) {      // two 32-deep sub-chunks
      BFrag a[2], b[2];
      #pragma unroll
      for (int mt = 0; mt < 2; ++mt) {
        const unsigned* base = &As[(wm + mt * 16 + l15) * LSTR + c * 16];
        #pragma unroll
        for (int j = 0; j < 4; ++j) {
          a[mt].u[j]     = base[half4 + j];
          a[mt].u[j + 4] = base[8 + half4 + j];
        }
      }
      #pragma unroll
      for (int nt = 0; nt < 2; ++nt) {
        const unsigned* base = &Ws[(wn + nt * 16 + l15) * LSTR + c * 16];
        #pragma unroll
        for (int j = 0; j < 8; ++j) b[nt].u[j] = base[half4 + (j & 3) + ((j >> 2) << 3)];
      }
      #pragma unroll
      for (int mt = 0; mt < 2; ++mt)
        #pragma unroll
        for (int nt = 0; nt < 2; ++nt)
          acc[mt][nt] = wmma_bf16(a[mt], b[nt], acc[mt][nt]);
    }
  }

  // epilogue: C layout lane<16 -> rows r, lanes>=16 -> rows r+8; col = lane&15
  #pragma unroll
  for (int mt = 0; mt < 2; ++mt)
    #pragma unroll
    for (int nt = 0; nt < 2; ++nt) {
      int gcol = n0 + wn + nt * 16 + l15;
      #pragma unroll
      for (int r = 0; r < 8; ++r) {
        int grow = m0 + wm + mt * 16 + ((lane < 16) ? r : r + 8);
        size_t off = (size_t)grow * Nc + gcol;
        float val = acc[mt][nt][r];
        if (HAS_RESID) val += resid[off];
        if (OUT_BF16) ((unsigned short*)Cp)[off] = f2bf(val);
        else          ((float*)Cp)[off] = val;
      }
    }
}

// ---------------------------------------------------------------------------
// RMSNorm over rows of 1024; out f32 or bf16
// ---------------------------------------------------------------------------
__global__ __launch_bounds__(256)
void rmsnorm_k(const float* __restrict__ x, const float* __restrict__ w,
               void* __restrict__ out, int outBF16) {
  __shared__ float red[8];
  int row = blockIdx.x, tid = threadIdx.x;
  const float* xr = x + (size_t)row * DMODEL;
  float v[4], s = 0.f;
  #pragma unroll
  for (int j = 0; j < 4; ++j) { v[j] = xr[tid + j * 256]; s += v[j] * v[j]; }
  s = half16_sum(s);            // DPP within 16-lane rows
  s += __shfl_xor(s, 16, 32);   // cross-half step
  if ((tid & 31) == 0) red[tid >> 5] = s;
  __syncthreads();
  float tot = 0.f;
  #pragma unroll
  for (int i = 0; i < 8; ++i) tot += red[i];
  float rs = rsqrtf(tot * (1.0f / DMODEL) + 1e-5f);
  #pragma unroll
  for (int j = 0; j < 4; ++j) {
    int c = tid + j * 256;
    float o = v[j] * rs * w[c];
    if (outBF16) ((unsigned short*)out)[(size_t)row * DMODEL + c] = f2bf(o);
    else         ((float*)out)[(size_t)row * DMODEL + c] = o;
  }
}

// ---------------------------------------------------------------------------
// 2D RoPE on q,k (bf16, in-place). pair j: j<16 -> y*theta^(-2j/32),
// j>=16 -> x*theta^(-(2(j-16)+1)/32); rotates dims (2j, 2j+1) of each head.
// ---------------------------------------------------------------------------
__global__ void rope_k(unsigned short* __restrict__ qb, unsigned short* __restrict__ kb) {
  int i = blockIdx.x * blockDim.x + threadIdx.x;     // N*NH*32
  if (i >= NTOK * 16 * 32) return;
  int j = i & 31, h = (i >> 5) & 15, t = i >> 9;
  int loc, wp;
  if (t < 512)       { loc = t;        wp = 32; }
  else if (t < 1536) { loc = t - 512;  wp = 32; }
  else if (t < 2048) { loc = t - 1536; wp = 16; }
  else               { loc = t - 2048; wp = 16; }
  int y = loc / wp, x = loc % wp;
  int tt = (j < 16) ? (2 * j) : (2 * (j - 16) + 1);
  float coord = (j < 16) ? (float)y : (float)x;
  float ang = coord * __powf(10000.0f, -(float)tt * (1.0f / 32.0f));
  float s, c;
  __sincosf(ang, &s, &c);
  size_t off = (size_t)t * DMODEL + h * 64 + 2 * j;
  float qr = bf2f(qb[off]), qi = bf2f(qb[off + 1]);
  qb[off] = f2bf(qr * c - qi * s); qb[off + 1] = f2bf(qr * s + qi * c);
  float kr = bf2f(kb[off]), ki = bf2f(kb[off + 1]);
  kb[off] = f2bf(kr * c - ki * s); kb[off + 1] = f2bf(kr * s + ki * c);
}

// ---------------------------------------------------------------------------
// Flash attention, WMMA. grid=(36 query tiles of 64 rows, 16 heads),
// 128 threads = 4 waves, each wave owns 16 query rows. Block-diagonal mask
// handled by iterating keys only within the owning image (all lens %64==0).
// Online-softmax reductions use DPP16 (pure VALU, co-executes with WMMA).
// ---------------------------------------------------------------------------
__global__ __launch_bounds__(128)
void attn_k(const unsigned short* __restrict__ qb, const unsigned short* __restrict__ kb,
            const unsigned short* __restrict__ vb, unsigned short* __restrict__ ob) {
  int qt = blockIdx.x, h = blockIdx.y;
  int seqstart, seqlen, tin;
  if (qt < 8)       { seqstart = 0;    seqlen = 512;  tin = qt; }
  else if (qt < 24) { seqstart = 512;  seqlen = 1024; tin = qt - 8; }
  else if (qt < 32) { seqstart = 1536; seqlen = 512;  tin = qt - 24; }
  else              { seqstart = 2048; seqlen = 256;  tin = qt - 32; }
  int qbase = seqstart + tin * 64;

  __shared__ unsigned Ks[32 * 44];       // 32 keys x 64 dims (32 dw) + pad
  __shared__ unsigned Vt[64 * 20];       // 64 dims x 32 keys (16 dw) + pad
  __shared__ unsigned Ps[4 * 16 * 20];   // per-wave P: 16 rows x 32 (16 dw) + pad

  int tid = threadIdx.x, lane = tid & 31, wid = tid >> 5, l15 = lane & 15;
  int half4 = (lane >= 16) ? 4 : 0, half8 = (lane >= 16) ? 8 : 0;

  // Q fragments for k-dims 0-31 and 32-63 (A-layout, from global)
  BFrag qf[2];
  {
    int qrow = qbase + wid * 16 + l15;
    const unsigned* qp = (const unsigned*)qb + ((size_t)qrow * DMODEL + h * 64) / 2;
    #pragma unroll
    for (int c = 0; c < 2; ++c) {
      const unsigned* base = qp + c * 16;
      #pragma unroll
      for (int j = 0; j < 4; ++j) {
        qf[c].u[j]     = base[half4 + j];
        qf[c].u[j + 4] = base[8 + half4 + j];
      }
    }
  }

  v8f accO[4] = {};
  float mrow[8], lrow[8];
  #pragma unroll
  for (int r = 0; r < 8; ++r) { mrow[r] = -1e30f; lrow[r] = 0.f; }

  for (int kk = 0; kk < seqlen; kk += 32) {
    __syncthreads();
    // K chunk: 32 keys x 64 dims (row-major in LDS, stride 44 dwords)
    #pragma unroll
    for (int it = 0; it < 2; ++it) {
      int idx = tid + it * 128;
      int key = idx >> 3, seg = idx & 7;
      uint4 v = *((const uint4*)((const unsigned*)kb +
                 ((size_t)(seqstart + kk + key) * DMODEL + h * 64) / 2) + seg);
      *(uint4*)(&Ks[key * 44 + seg * 4]) = v;
    }
    // V chunk transposed: Vt[dim][key], stride 20 dwords
    {
      int key = tid & 31, g = tid >> 5;
      const unsigned* src = (const unsigned*)vb +
          ((size_t)(seqstart + kk + key) * DMODEL + h * 64 + g * 16) / 2;
      unsigned short* vt16 = (unsigned short*)Vt;
      #pragma unroll
      for (int j = 0; j < 8; ++j) {
        unsigned d = src[j];
        int dim0 = g * 16 + 2 * j;
        vt16[dim0 * 40 + key]       = (unsigned short)(d & 0xffffu);
        vt16[(dim0 + 1) * 40 + key] = (unsigned short)(d >> 16);
      }
    }
    __syncthreads();

    // S = Q * K^T : two 16x16 tiles (keys 0-15 and 16-31)
    v8f S0 = {}, S1 = {};
    #pragma unroll
    for (int c = 0; c < 2; ++c) {
      BFrag b0, b1;
      #pragma unroll
      for (int j = 0; j < 8; ++j) {
        b0.u[j] = Ks[l15 * 44 + c * 16 + half4 + (j & 3) + ((j >> 2) << 3)];
        b1.u[j] = Ks[(16 + l15) * 44 + c * 16 + half4 + (j & 3) + ((j >> 2) << 3)];
      }
      S0 = wmma_bf16(qf[c], b0, S0);
      S1 = wmma_bf16(qf[c], b1, S1);
    }

    // online softmax; P staged into per-wave LDS as bf16
    unsigned short* P = (unsigned short*)&Ps[wid * 16 * 20];
    #pragma unroll
    for (int r = 0; r < 8; ++r) {
      float s0 = S0[r] * 0.125f, s1 = S1[r] * 0.125f;
      float mx = half16_max(fmaxf(s0, s1));
      float mnew = fmaxf(mrow[r], mx);
      float alpha = __expf(mrow[r] - mnew);
      float p0 = __expf(s0 - mnew), p1 = __expf(s1 - mnew);
      float ps = half16_sum(p0 + p1);
      lrow[r] = lrow[r] * alpha + ps;
      mrow[r] = mnew;
      #pragma unroll
      for (int g = 0; g < 4; ++g) accO[g][r] *= alpha;
      int prow = (lane < 16) ? r : r + 8;
      P[prow * 40 + l15]      = f2bf(p0);
      P[prow * 40 + 16 + l15] = f2bf(p1);
    }

    // O += P * V   (per-wave LDS; no barrier needed, DScnt keeps order)
    BFrag pf;
    {
      const unsigned* pb = &Ps[wid * 16 * 20] + l15 * 20;
      #pragma unroll
      for (int j = 0; j < 4; ++j) {
        pf.u[j]     = pb[half4 + j];
        pf.u[j + 4] = pb[8 + half4 + j];
      }
    }
    #pragma unroll
    for (int g = 0; g < 4; ++g) {
      BFrag bv;
      #pragma unroll
      for (int j = 0; j < 8; ++j) bv.u[j] = Vt[(g * 16 + l15) * 20 + half4 + (j & 3) + ((j >> 2) << 3)];
      accO[g] = wmma_bf16(pf, bv, accO[g]);
    }
  }

  // normalize + write bf16
  #pragma unroll
  for (int g = 0; g < 4; ++g)
    #pragma unroll
    for (int r = 0; r < 8; ++r) {
      int row = qbase + wid * 16 + ((lane < 16) ? r : r + 8);
      int col = h * 64 + g * 16 + l15;
      ob[(size_t)row * DMODEL + col] = f2bf(accO[g][r] / lrow[r]);
    }
}

// ---------------------------------------------------------------------------
// SwiGLU gate: hm = bf16( silu(t1) * t3 )
// ---------------------------------------------------------------------------
__global__ void silumul_k(const float* __restrict__ t1, const float* __restrict__ t3,
                          unsigned short* __restrict__ hm, int n) {
  int i = blockIdx.x * blockDim.x + threadIdx.x;
  if (i >= n) return;
  float a = t1[i];
  hm[i] = f2bf((a / (1.0f + __expf(-a))) * t3[i]);
}

// ---------------------------------------------------------------------------
// Host orchestration
// ---------------------------------------------------------------------------
extern "C" void kernel_launch(void* const* d_in, const int* in_sizes, int n_in,
                              void* d_out, int out_size, void* d_ws, size_t ws_size,
                              hipStream_t stream) {
  const float* img0    = (const float*)d_in[0];
  const float* img1    = (const float*)d_in[1];
  const float* img2    = (const float*)d_in[2];
  const float* img3    = (const float*)d_in[3];
  const float* patch_w = (const float*)d_in[4];   // (1024, 768)
  const float* ln_pre  = (const float*)d_in[5];
  const float* wq      = (const float*)d_in[6];   // (L, D, D)
  const float* wk      = (const float*)d_in[7];
  const float* wv      = (const float*)d_in[8];
  const float* wo      = (const float*)d_in[9];
  const float* w1      = (const float*)d_in[10];  // (L, DFF, D)
  const float* w2      = (const float*)d_in[11];  // (L, D, DFF)
  const float* w3      = (const float*)d_in[12];  // (L, DFF, D)
  const float* an_w    = (const float*)d_in[13];  // (L, D)
  const float* fn_w    = (const float*)d_in[14];  // (L, D)
  (void)in_sizes; (void)n_in; (void)out_size; (void)ws_size;

  size_t off = 0;
  auto alloc = [&](size_t bytes) -> void* {
    void* p = (char*)d_ws + off;
    off = (off + bytes + 255) & ~(size_t)255;
    return p;
  };
  float*          x   = (float*)alloc((size_t)NTOK * DMODEL * 4);
  float*          t1  = (float*)alloc((size_t)NTOK * DFF * 4);
  float*          t3  = (float*)alloc((size_t)NTOK * DFF * 4);
  unsigned short* hb  = (unsigned short*)alloc((size_t)NTOK * DMODEL * 2);
  unsigned short* qb  = (unsigned short*)alloc((size_t)NTOK * DMODEL * 2);
  unsigned short* kbp = (unsigned short*)alloc((size_t)NTOK * DMODEL * 2);
  unsigned short* vbp = (unsigned short*)alloc((size_t)NTOK * DMODEL * 2);
  unsigned short* ob  = (unsigned short*)alloc((size_t)NTOK * DMODEL * 2);
  unsigned short* hm  = (unsigned short*)alloc((size_t)NTOK * DFF * 2);
  unsigned short* imb = (unsigned short*)alloc((size_t)NTOK * 768 * 2);

  dim3 gD(DMODEL / 64, NTOK / 64);   // (16, 36)
  dim3 gF(DFF / 64, NTOK / 64);      // (64, 36)

  // patch embed: im2col -> GEMM (K=768) -> rmsnorm(ln_pre) into x
  im2col_k<<<(NTOK * 768 + 255) / 256, 256, 0, stream>>>(img0, img1, img2, img3, imb);
  gemm_bf16_wmma<0, 0><<<gD, 128, 0, stream>>>(imb, patch_w, nullptr, t1, NTOK, DMODEL, 768);
  rmsnorm_k<<<NTOK, 256, 0, stream>>>(t1, ln_pre, x, 0);

  for (int l = 0; l < 4; ++l) {
    const float* wql = wq + (size_t)l * DMODEL * DMODEL;
    const float* wkl = wk + (size_t)l * DMODEL * DMODEL;
    const float* wvl = wv + (size_t)l * DMODEL * DMODEL;
    const float* wol = wo + (size_t)l * DMODEL * DMODEL;
    const float* w1l = w1 + (size_t)l * DFF * DMODEL;
    const float* w2l = w2 + (size_t)l * DMODEL * DFF;
    const float* w3l = w3 + (size_t)l * DFF * DMODEL;

    rmsnorm_k<<<NTOK, 256, 0, stream>>>(x, an_w + (size_t)l * DMODEL, hb, 1);
    gemm_bf16_wmma<1, 0><<<gD, 128, 0, stream>>>(hb, wql, nullptr, qb, NTOK, DMODEL, DMODEL);
    gemm_bf16_wmma<1, 0><<<gD, 128, 0, stream>>>(hb, wkl, nullptr, kbp, NTOK, DMODEL, DMODEL);
    gemm_bf16_wmma<1, 0><<<gD, 128, 0, stream>>>(hb, wvl, nullptr, vbp, NTOK, DMODEL, DMODEL);
    rope_k<<<(NTOK * 16 * 32 + 255) / 256, 256, 0, stream>>>(qb, kbp);
    attn_k<<<dim3(36, 16), 128, 0, stream>>>(qb, kbp, vbp, ob);
    gemm_bf16_wmma<0, 1><<<gD, 128, 0, stream>>>(ob, wol, x, x, NTOK, DMODEL, DMODEL);

    rmsnorm_k<<<NTOK, 256, 0, stream>>>(x, fn_w + (size_t)l * DMODEL, hb, 1);
    gemm_bf16_wmma<0, 0><<<gF, 128, 0, stream>>>(hb, w1l, nullptr, t1, NTOK, DFF, DMODEL);
    gemm_bf16_wmma<0, 0><<<gF, 128, 0, stream>>>(hb, w3l, nullptr, t3, NTOK, DFF, DMODEL);
    silumul_k<<<(NTOK * DFF + 255) / 256, 256, 0, stream>>>(t1, t3, hm, NTOK * DFF);
    float* outp = (l == 3) ? (float*)d_out : x;
    gemm_bf16_wmma<0, 1><<<gD, 128, 0, stream>>>(hm, w2l, x, outp, NTOK, DMODEL, DFF);
  }
}
